// PerformerAttention_29016799052236
// MI455X (gfx1250) — matrix operand order repaired and checked by reading, not compiled, not run on hardware
//
#include <hip/hip_runtime.h>

// ---------------------------------------------------------------------------
// Performer-style attention for MI455X (gfx1250), wave32 + WMMA bf16 path.
// Pipeline: f32->bf16 convert, 3 projection GEMMs (relu/scale epilogues),
// per-token 64x64x64 WMMA attention, output GEMM with bias (f32 out).
// GEMM: 128x128 block tile, 8 waves, each wave = 32x64 (2x4 WMMA accs),
// so each staged LDS fragment feeds 4 (A) or 2 (B) wmma ops.
// ---------------------------------------------------------------------------

typedef __attribute__((ext_vector_type(16))) __bf16 v16bf;
typedef __attribute__((ext_vector_type(8)))  __bf16 v8bf;
typedef __attribute__((ext_vector_type(8)))  float  v8f;

#define HEADS    64
#define DIM_HEAD 64
#define P_SCALE  0.125f   // DIM_HEAD^-0.5
#define P_EPS    1e-6f

// ---------------------------------------------------------------------------
// f32 -> bf16 conversion, 8 elements per thread (b128 load / b128 store)
// ---------------------------------------------------------------------------
__global__ void cvt_f32_bf16(const float* __restrict__ s, __bf16* __restrict__ d, int n) {
    int i = (blockIdx.x * blockDim.x + threadIdx.x) * 8;
    if (i + 8 <= n) {
#pragma unroll
        for (int e = 0; e < 8; ++e) d[i + e] = (__bf16)s[i + e];
    } else {
        for (; i < n; ++i) d[i] = (__bf16)s[i];
    }
}

// ---------------------------------------------------------------------------
// Tiled bf16 GEMM:  C[M,N] = epilogue( A[M,K] @ B[K,N] )
// Block: 256 threads (8 waves). Block tile 128x128, K-step 32.
// Waves arranged 4x2; each wave computes 32x64 = 2x4 grid of 16x16 accs.
// A staged to LDS row-major [m][k]; B staged transposed [n][k] so both
// fragments are contiguous 16B LDS reads matching the bf16 WMMA VGPR layout:
//   A lane L: row = L&15, K chunks {h*8..h*8+7, 16+h*8..} (h = L>>4)
//   B lane L: col = L&15, K chunks {h*8..h*8+7, 16+h*8..} (h = L>>4)
// C layout: lane L, vgpr r -> (m = (L>>4)*8 + r, n = L&15)
// ---------------------------------------------------------------------------
template <bool RELU, bool BIAS, typename OutT>
__global__ __launch_bounds__(256) void gemm_wmma_bf16(
        const __bf16* __restrict__ A,
        const __bf16* __restrict__ B,
        OutT* __restrict__ C,
        const float* __restrict__ bias,
        int M, int N, int K, float alpha) {
    constexpr int BM = 128, BN = 128, KS = 32;
    __shared__ __attribute__((aligned(16))) __bf16 sA[BM * KS];   // [m][k]
    __shared__ __attribute__((aligned(16))) __bf16 sBt[BN * KS];  // [n][k]

    const int tid  = threadIdx.x;
    const int lane = tid & 31;
    const int wave = tid >> 5;       // 0..7
    const int wm   = wave >> 1;      // 0..3 -> 32-row strip
    const int wn   = wave & 1;       // 0..1 -> 64-col strip
    const int bm = blockIdx.y * BM;
    const int bn = blockIdx.x * BN;

    const int lm   = lane & 15;
    const int half = lane >> 4;

    // cooperative staging coordinates (256 threads, 16 bf16 each per tile)
    const int ar = tid >> 1;          // 0..127 : A-tile row
    const int ac = (tid & 1) * 16;    // A-tile k group
    const int br = tid >> 3;          // 0..31  : B-tile k row
    const int bc = (tid & 7) * 16;    // B-tile n group

    v8f zero = {};
    v8f acc[2][4];
#pragma unroll
    for (int mt = 0; mt < 2; ++mt)
#pragma unroll
        for (int nt = 0; nt < 4; ++nt) acc[mt][nt] = zero;

    for (int k0 = 0; k0 < K; k0 += KS) {
        // ---- stage A tile (two b128 per thread) ----
        {
            const __bf16* g = A + (size_t)(bm + ar) * K + k0 + ac;
            v8bf a0 = *(const v8bf*)(g);
            v8bf a1 = *(const v8bf*)(g + 8);
            *(v8bf*)(sA + ar * KS + ac)     = a0;
            *(v8bf*)(sA + ar * KS + ac + 8) = a1;
        }
        // ---- stage B tile, transposed into [n][k] ----
        {
            const __bf16* g = B + (size_t)(k0 + br) * N + bn + bc;
            v8bf b0 = *(const v8bf*)(g);
            v8bf b1 = *(const v8bf*)(g + 8);
#pragma unroll
            for (int e = 0; e < 8; ++e) {
                sBt[(bc + e) * KS + br]     = b0[e];
                sBt[(bc + 8 + e) * KS + br] = b1[e];
            }
        }
        __syncthreads();

        if (k0 + KS < K) {  // hint next tiles into cache (global_prefetch_b8)
            __builtin_prefetch(A + (size_t)(bm + ar) * K + k0 + KS + ac, 0, 3);
            __builtin_prefetch(B + (size_t)(k0 + KS + br) * N + bn + bc, 0, 3);
        }

        // ---- build fragments ----
        v16bf af[2];
#pragma unroll
        for (int mt = 0; mt < 2; ++mt) {
            const __bf16* p = sA + (wm * 32 + mt * 16 + lm) * KS;
            v8bf c0 = *(const v8bf*)(p + half * 8);
            v8bf c1 = *(const v8bf*)(p + 16 + half * 8);
#pragma unroll
            for (int e = 0; e < 8; ++e) { af[mt][e] = c0[e]; af[mt][8 + e] = c1[e]; }
        }
        v16bf bfr[4];
#pragma unroll
        for (int nt = 0; nt < 4; ++nt) {
            const __bf16* p = sBt + (wn * 64 + nt * 16 + lm) * KS + half * 8;
            v8bf c0 = *(const v8bf*)(p);
            v8bf c1 = *(const v8bf*)(p + 16);
#pragma unroll
            for (int e = 0; e < 8; ++e) { bfr[nt][e] = c0[e]; bfr[nt][8 + e] = c1[e]; }
        }
        // ---- 8 WMMAs per wave per K-step ----
#pragma unroll
        for (int mt = 0; mt < 2; ++mt)
#pragma unroll
            for (int nt = 0; nt < 4; ++nt)
                acc[mt][nt] = __builtin_amdgcn_wmma_f32_16x16x32_bf16(
                        false, af[mt], false, bfr[nt], (short)0, acc[mt][nt], false, false);
        __syncthreads();
    }

    // ---- epilogue ----
#pragma unroll
    for (int mt = 0; mt < 2; ++mt) {
        const int mrow = bm + wm * 32 + mt * 16 + half * 8;
#pragma unroll
        for (int nt = 0; nt < 4; ++nt) {
            const int ncol = bn + wn * 64 + nt * 16 + lm;
            float bb = 0.f;
            if (BIAS) bb = bias[ncol];
#pragma unroll
            for (int r = 0; r < 8; ++r) {
                float x = acc[mt][nt][r];
                if (RELU) x = fmaxf(x, 0.f);
                x = x * alpha + bb;
                C[(size_t)(mrow + r) * N + ncol] = (OutT)x;
            }
        }
    }
}

// ---------------------------------------------------------------------------
// Per-token kernel. One block (128 threads, 4 waves) per token:
//   ksum[d]   = sum_h k[h,d]
//   qninv[i]  = 1 / (sum_d q[i,d]*ksum[d] + eps)
//   out[i,j]  = (sum_d q[i,d]*ksum[d]*v[j,d]) * qninv[i]
// A-fragment = q scaled by ksum (built in registers); B-fragment = v^T,
// which is lane-contiguous in memory since v is stored [head][dim].
// Wave w computes rows i0=w*16..w*16+15, all 64 columns (4 accs x 2 K-steps).
// ---------------------------------------------------------------------------
__global__ __launch_bounds__(128) void performer_token(
        const __bf16* __restrict__ Q,
        const __bf16* __restrict__ Km,
        const __bf16* __restrict__ V,
        __bf16* __restrict__ Out) {
    const size_t base = (size_t)blockIdx.x * (HEADS * DIM_HEAD);
    __shared__ float part[2][DIM_HEAD];
    __shared__ float ksum[DIM_HEAD];
    __shared__ float qninv[HEADS];

    const int tid  = threadIdx.x;    // 0..127
    const int lane = tid & 31;
    const int wave = tid >> 5;       // 0..3

    // ---- ksum over heads ----
    {
        const int d  = tid & 63;
        const int hb = (tid >> 6) * 32;
        float s = 0.f;
        for (int h = 0; h < 32; ++h)
            s += (float)Km[base + (size_t)(hb + h) * DIM_HEAD + d];
        part[tid >> 6][d] = s;
    }
    __syncthreads();
    if (tid < DIM_HEAD) ksum[tid] = part[0][tid] + part[1][tid];
    __syncthreads();

    // ---- 1/(q_norm + eps) per output row ----
    if (tid < HEADS) {
        float s = 0.f;
        for (int d = 0; d < DIM_HEAD; ++d)
            s += (float)Q[base + (size_t)tid * DIM_HEAD + d] * ksum[d];
        qninv[tid] = 1.f / (s + P_EPS);
    }
    __syncthreads();

    const int lm   = lane & 15;
    const int half = lane >> 4;
    const int i0   = wave * 16;

    v8f zero = {};
    v8f acc[4] = {zero, zero, zero, zero};

#pragma unroll
    for (int ks = 0; ks < 2; ++ks) {
        // A fragment: a[i,d] = q[i,d] * ksum[d]
        v16bf afrag;
        {
            const int m  = i0 + lm;
            const int d0 = ks * 32 + half * 8;
            const int d1 = d0 + 16;
            v8bf q0 = *(const v8bf*)(Q + base + (size_t)m * DIM_HEAD + d0);
            v8bf q1 = *(const v8bf*)(Q + base + (size_t)m * DIM_HEAD + d1);
#pragma unroll
            for (int e = 0; e < 8; ++e) {
                afrag[e]     = (__bf16)((float)q0[e] * ksum[d0 + e]);
                afrag[8 + e] = (__bf16)((float)q1[e] * ksum[d1 + e]);
            }
        }
#pragma unroll
        for (int jt = 0; jt < 4; ++jt) {
            const int j  = jt * 16 + lm;
            const int d0 = ks * 32 + half * 8;
            v8bf v0 = *(const v8bf*)(V + base + (size_t)j * DIM_HEAD + d0);
            v8bf v1 = *(const v8bf*)(V + base + (size_t)j * DIM_HEAD + d0 + 16);
            v16bf bfrag;
#pragma unroll
            for (int e = 0; e < 8; ++e) { bfrag[e] = v0[e]; bfrag[8 + e] = v1[e]; }
            acc[jt] = __builtin_amdgcn_wmma_f32_16x16x32_bf16(false, afrag, false, bfrag,
                                                              (short)0, acc[jt], false, false);
        }
    }

    // ---- scale rows by qninv, store bf16 ----
#pragma unroll
    for (int jt = 0; jt < 4; ++jt) {
#pragma unroll
        for (int r = 0; r < 8; ++r) {
            const int i = i0 + half * 8 + r;
            const int j = jt * 16 + lm;
            Out[base + (size_t)i * HEADS + j] = (__bf16)(acc[jt][r] * qninv[i]);
        }
    }
}

// ---------------------------------------------------------------------------
// host launcher
// ---------------------------------------------------------------------------
extern "C" void kernel_launch(void* const* d_in, const int* in_sizes, int n_in,
                              void* d_out, int out_size, void* d_ws, size_t ws_size,
                              hipStream_t stream) {
    (void)in_sizes; (void)n_in; (void)out_size; (void)ws_size;
    const float* x  = (const float*)d_in[0];
    const float* Wq = (const float*)d_in[1];
    const float* Wk = (const float*)d_in[2];
    const float* Wv = (const float*)d_in[3];
    const float* Wo = (const float*)d_in[4];
    const float* bo = (const float*)d_in[5];
    float* out = (float*)d_out;

    const int DIMV  = 1024;
    const int INNER = HEADS * DIM_HEAD;   // 4096
    const int M     = 2 * 4096;           // b*s = 8192

    // workspace layout (bf16 buffers)
    char* ws = (char*)d_ws;
    __bf16* xh   = (__bf16*)ws; ws += (size_t)M * DIMV * 2;
    __bf16* wqh  = (__bf16*)ws; ws += (size_t)DIMV * INNER * 2;
    __bf16* wkh  = (__bf16*)ws; ws += (size_t)DIMV * INNER * 2;
    __bf16* wvh  = (__bf16*)ws; ws += (size_t)DIMV * INNER * 2;
    __bf16* woh  = (__bf16*)ws; ws += (size_t)INNER * DIMV * 2;
    __bf16* qh   = (__bf16*)ws; ws += (size_t)M * INNER * 2;
    __bf16* kh   = (__bf16*)ws; ws += (size_t)M * INNER * 2;
    __bf16* vh   = (__bf16*)ws; ws += (size_t)M * INNER * 2;
    __bf16* atth = (__bf16*)ws; ws += (size_t)M * INNER * 2;

    auto cvt = [&](const float* s, __bf16* d, int n) {
        int threads = n / 8;
        cvt_f32_bf16<<<(threads + 255) / 256, 256, 0, stream>>>(s, d, n);
    };
    cvt(x,  xh,  M * DIMV);
    cvt(Wq, wqh, DIMV * INNER);
    cvt(Wk, wkh, DIMV * INNER);
    cvt(Wv, wvh, DIMV * INNER);
    cvt(Wo, woh, INNER * DIMV);

    // projections: q = relu(x@Wq)*SCALE, k = relu(x@Wk), v = x@Wv
    dim3 gp(INNER / 128, M / 128);
    gemm_wmma_bf16<true,  false, __bf16><<<gp, 256, 0, stream>>>(xh, wqh, qh, nullptr, M, INNER, DIMV, P_SCALE);
    gemm_wmma_bf16<true,  false, __bf16><<<gp, 256, 0, stream>>>(xh, wkh, kh, nullptr, M, INNER, DIMV, 1.0f);
    gemm_wmma_bf16<false, false, __bf16><<<gp, 256, 0, stream>>>(xh, wvh, vh, nullptr, M, INNER, DIMV, 1.0f);

    // per-token attention (64x64x64 WMMA per token)
    performer_token<<<M, 128, 0, stream>>>(qh, kh, vh, atth);

    // output projection: out = att @ Wo + bo (f32 out)
    dim3 go(DIMV / 128, M / 128);
    gemm_wmma_bf16<false, true, float><<<go, 256, 0, stream>>>(atth, woh, out, bo, M, DIMV, INNER, 1.0f);
}
